// BeliefGNN_36704790511854
// MI455X (gfx1250) — matrix operand order repaired and checked by reading, not compiled, not run on hardware
//
#include <hip/hip_runtime.h>
#include <hip/hip_bf16.h>

typedef __attribute__((ext_vector_type(16))) __bf16 v16bf;
typedef __attribute__((ext_vector_type(8)))  float  v8f;

// round-to-nearest-even f32 -> bf16
__device__ __forceinline__ unsigned short f32_bf16(float f) {
    union { float f; unsigned int u; } c; c.f = f;
    unsigned int u = c.u;
    unsigned int r = 0x7FFFu + ((u >> 16) & 1u);
    return (unsigned short)((u + r) >> 16);
}

// A-operand (16-bit 16x32, ISA 7.12.2): lanes 0-15 row M=m0+lane, VGPR0-3 K=k0..k0+7,
// VGPR4-7 K=k0+16..k0+23; lanes 16-31 same rows, K offset +8. kx=64 swaps the
// two 64-wide halves of the row (for the [h_j|h_i] pass).
__device__ __forceinline__ v16bf load_a(const unsigned short* __restrict__ s,
                                        int m0, int k0, int lane, int kx) {
    int m    = m0 + (lane & 15);
    int half = lane >> 4;
    const unsigned short* row = s + m * 128;
    int c0 = (k0 + half * 8) ^ kx;
    int c1 = (k0 + 16 + half * 8) ^ kx;
    union { uint4 u[2]; v16bf v; } t;
    t.u[0] = *(const uint4*)(row + c0);
    t.u[1] = *(const uint4*)(row + c1);
    return t.v;
}

// B-operand from transposed weights sBt[N][128]: lane n = n0+(lane&15),
// lanes 0-15 hold K=k0..k0+7,k0+16..k0+23; lanes 16-31 K offset +8.
__device__ __forceinline__ v16bf load_b(const unsigned short* __restrict__ s,
                                        int n0, int k0, int lane) {
    int n    = n0 + (lane & 15);
    int half = lane >> 4;
    const unsigned short* p = s + n * 128 + k0 + half * 8;
    union { uint4 u[2]; v16bf v; } t;
    t.u[0] = *(const uint4*)(p);
    t.u[1] = *(const uint4*)(p + 16);
    return t.v;
}

__global__ void __launch_bounds__(256)
gnn_init_copy(float* __restrict__ out, const float* __restrict__ src, int n) {
    int i = blockIdx.x * 256 + threadIdx.x;
    if (i < n) out[i] = src[i];
}

// one-time: W1 (128x128) and W2 (128x64) -> bf16, transposed [N][K], into ws
__global__ void __launch_bounds__(256)
gnn_cvt_weights(const float* __restrict__ W1, const float* __restrict__ W2,
                unsigned short* __restrict__ wsW1t, unsigned short* __restrict__ wsW2t) {
    int idx = blockIdx.x * 256 + threadIdx.x;
    if (idx < 128 * 128) {
        int k = idx >> 7, n = idx & 127;
        wsW1t[n * 128 + k] = f32_bf16(W1[idx]);
    } else if (idx < 128 * 128 + 128 * 64) {
        int j = idx - 128 * 128;
        int k = j >> 6, n = j & 63;
        wsW2t[n * 128 + k] = f32_bf16(W2[j]);
    }
}

// one-time: node features -> bf16 copy in ws (L2-resident gather source)
__global__ void __launch_bounds__(256)
gnn_cvt_nodes(const float* __restrict__ nodes, unsigned short* __restrict__ dst, int n) {
    int i = blockIdx.x * 256 + threadIdx.x;
    if (i < n) dst[i] = f32_bf16(nodes[i]);
}

// MODE 0: convert weights + gather from f32 in-block (no ws)
// MODE 1: weights raw-copied from ws; gather converts f32 nodes
// MODE 2: weights raw-copied from ws; gather raw-copies bf16 nodes from ws
template <int MODE>
__global__ void __launch_bounds__(256)
gnn_edges(const float* __restrict__ nodes, const unsigned short* __restrict__ nodesB,
          const int* __restrict__ edges,
          const float* __restrict__ W1, const float* __restrict__ b1,
          const float* __restrict__ W2, const float* __restrict__ b2,
          const unsigned short* __restrict__ wsW1t, const unsigned short* __restrict__ wsW2t,
          float* __restrict__ out, int E) {
    __shared__ __align__(16) unsigned short sW1t[128 * 128]; // [N][K] bf16
    __shared__ __align__(16) unsigned short sW2t[64 * 128];  // [N][K] bf16
    __shared__ __align__(16) unsigned short sX[128 * 128];   // 128 edges x [h_i|h_j]
    __shared__ __align__(16) unsigned short sY[128 * 128];   // hidden activations
    __shared__ float sB1[128];
    __shared__ float sB2[64];
    __shared__ int   sIdx[256];                              // (i,j) per edge, -1 = pad

    const int tid  = threadIdx.x;
    const int lane = tid & 31;
    const int wave = tid >> 5;

    // ---- weights -> LDS ----
    if constexpr (MODE == 0) {
        for (int idx = tid; idx < (128 * 128) / 4; idx += 256) {
            float4 f = ((const float4*)W1)[idx];
            int lin = idx << 2;
            int k = lin >> 7, n = lin & 127;
            sW1t[(n + 0) * 128 + k] = f32_bf16(f.x);
            sW1t[(n + 1) * 128 + k] = f32_bf16(f.y);
            sW1t[(n + 2) * 128 + k] = f32_bf16(f.z);
            sW1t[(n + 3) * 128 + k] = f32_bf16(f.w);
        }
        for (int idx = tid; idx < (128 * 64) / 4; idx += 256) {
            float4 f = ((const float4*)W2)[idx];
            int lin = idx << 2;
            int k = lin >> 6, n = lin & 63;
            sW2t[(n + 0) * 128 + k] = f32_bf16(f.x);
            sW2t[(n + 1) * 128 + k] = f32_bf16(f.y);
            sW2t[(n + 2) * 128 + k] = f32_bf16(f.z);
            sW2t[(n + 3) * 128 + k] = f32_bf16(f.w);
        }
    } else {
        // raw 48 KB copy of pre-transposed bf16 weights (L2-resident)
        const uint4* s1 = (const uint4*)wsW1t;
        uint4*       d1 = (uint4*)sW1t;
        for (int idx = tid; idx < 2048; idx += 256) d1[idx] = s1[idx];
        const uint4* s2 = (const uint4*)wsW2t;
        uint4*       d2 = (uint4*)sW2t;
        for (int idx = tid; idx < 1024; idx += 256) d2[idx] = s2[idx];
    }
    if (tid < 128) sB1[tid] = b1[tid];
    if (tid < 64)  sB2[tid] = b2[tid];

    // ---- gather: thread (2e+half) copies one node row into X ----
    {
        int e    = tid >> 1;
        int half = tid & 1;
        long gE  = (long)blockIdx.x * 128 + e;
        int node = 0, tag = -1;
        if (gE < (long)E) { node = edges[2 * gE + half]; tag = node; }
        sIdx[2 * e + half] = tag;
        if constexpr (MODE == 2) {
            const uint4* src = (const uint4*)(nodesB + (size_t)node * 64); // 8 x 16B
            uint4*       dst = (uint4*)(sX + e * 128 + half * 64);
#pragma unroll
            for (int c = 0; c < 8; ++c) dst[c] = src[c];
        } else {
            const float4* src = (const float4*)(nodes + (size_t)node * 64);
            unsigned short* dst = sX + e * 128 + half * 64;
#pragma unroll
            for (int c = 0; c < 16; ++c) {
                float4 f = src[c];
                uint2 p;
                p.x = (unsigned)f32_bf16(f.x) | ((unsigned)f32_bf16(f.y) << 16);
                p.y = (unsigned)f32_bf16(f.z) | ((unsigned)f32_bf16(f.w) << 16);
                *(uint2*)(dst + c * 4) = p;
            }
        }
    }

    __syncthreads(); // weights/bias were the only cross-wave writes; below is wave-local

    const int m0 = wave * 16; // each wave owns 16 edge rows

    for (int pass = 0; pass < 2; ++pass) {
        const int kx = pass << 6; // XOR 64: swap h_i / h_j halves on pass 1

        // ---- phase 1: Y = relu(X @ W1 + b1), 8 N-tiles x 4 K-steps ----
        v16bf a[4];
#pragma unroll
        for (int kt = 0; kt < 4; ++kt) a[kt] = load_a(sX, m0, kt * 32, lane, kx);
#pragma unroll
        for (int nt = 0; nt < 8; ++nt) {
            v8f acc = {};
#pragma unroll
            for (int kt = 0; kt < 4; ++kt) {
                v16bf bm = load_b(sW1t, nt * 16, kt * 32, lane);
                acc = __builtin_amdgcn_wmma_f32_16x16x32_bf16(
                    false, a[kt], false, bm, (short)0, acc, false, false);
            }
            int   n    = nt * 16 + (lane & 15);
            float bias = sB1[n];
            int   mA   = (lane >> 4) << 3;
#pragma unroll
            for (int r = 0; r < 8; ++r) {
                float v = fmaxf(acc[r] + bias, 0.0f);
                sY[(m0 + mA + r) * 128 + n] = f32_bf16(v);
            }
        }

        // ---- phase 2: M = Y @ W2 + b2, scatter-add (same wave wrote these Y rows) ----
        v16bf ya[4];
#pragma unroll
        for (int kt = 0; kt < 4; ++kt) ya[kt] = load_a(sY, m0, kt * 32, lane, 0);
#pragma unroll
        for (int nt = 0; nt < 4; ++nt) {
            v8f acc = {};
#pragma unroll
            for (int kt = 0; kt < 4; ++kt) {
                v16bf bm = load_b(sW2t, nt * 16, kt * 32, lane);
                acc = __builtin_amdgcn_wmma_f32_16x16x32_bf16(
                    false, ya[kt], false, bm, (short)0, acc, false, false);
            }
            int   n    = nt * 16 + (lane & 15);
            float bias = sB2[n];
            int   mA   = (lane >> 4) << 3;
#pragma unroll
            for (int r = 0; r < 8; ++r) {
                int row  = m0 + mA + r;
                int node = sIdx[2 * row + pass]; // pass0 -> i, pass1 -> j
                if (node >= 0) {
                    float v  = acc[r] + bias;
                    float* p = out + (size_t)node * 64 + n;
                    asm volatile("global_atomic_add_f32 %0, %1, off"
                                 :: "v"(p), "v"(v) : "memory");
                }
            }
        }
        // no barrier needed: per-wave-private sY rows, LDS ops in-order per wave
    }
}

extern "C" void kernel_launch(void* const* d_in, const int* in_sizes, int n_in,
                              void* d_out, int out_size, void* d_ws, size_t ws_size,
                              hipStream_t stream) {
    const float* nodes = (const float*)d_in[0];
    const int*   edges = (const int*)d_in[1];
    const float* W1    = (const float*)d_in[2];
    const float* b1    = (const float*)d_in[3];
    const float* W2    = (const float*)d_in[4];
    const float* b2    = (const float*)d_in[5];
    float* out = (float*)d_out;

    const int E      = in_sizes[1] / 2;
    const int nNodeE = in_sizes[0];              // N * 64 elements
    const int nb     = (E + 127) / 128;

    const size_t wW = (size_t)(128 * 128 + 64 * 128) * 2; // 48 KB bf16 weights
    const size_t wN = (size_t)nNodeE * 2;                 // bf16 node table

    int mode = 0;
    unsigned short *wsNodes = nullptr, *wsW1t = nullptr, *wsW2t = nullptr;
    if (ws_size >= wN + wW) {
        mode    = 2;
        wsNodes = (unsigned short*)d_ws;
        wsW1t   = wsNodes + nNodeE;
        wsW2t   = wsW1t + 128 * 128;
    } else if (ws_size >= wW) {
        mode  = 1;
        wsW1t = (unsigned short*)d_ws;
        wsW2t = wsW1t + 128 * 128;
    }

    gnn_init_copy<<<(out_size + 255) / 256, 256, 0, stream>>>(out, nodes, out_size);

    if (mode >= 1)
        gnn_cvt_weights<<<(128 * 128 + 128 * 64 + 255) / 256, 256, 0, stream>>>(W1, W2, wsW1t, wsW2t);
    if (mode == 2)
        gnn_cvt_nodes<<<(nNodeE + 255) / 256, 256, 0, stream>>>(nodes, wsNodes, nNodeE);

    switch (mode) {
    case 2:
        gnn_edges<2><<<nb, 256, 0, stream>>>(nodes, wsNodes, edges, W1, b1, W2, b2,
                                             wsW1t, wsW2t, out, E);
        break;
    case 1:
        gnn_edges<1><<<nb, 256, 0, stream>>>(nodes, nullptr, edges, W1, b1, W2, b2,
                                             wsW1t, wsW2t, out, E);
        break;
    default:
        gnn_edges<0><<<nb, 256, 0, stream>>>(nodes, nullptr, edges, W1, b1, W2, b2,
                                             nullptr, nullptr, out, E);
        break;
    }
}
